// Cross_SA_Layer_19550691131524
// MI455X (gfx1250) — matrix operand order repaired and checked by reading, not compiled, not run on hardware
//
#include <hip/hip_runtime.h>
#include <hip/hip_bf16.h>

// ---------------------------------------------------------------------------
// Problem constants (from reference): GROUP=4, SG=4, C=256, N=1024, BS=16
// ---------------------------------------------------------------------------
#define C_   256
#define N_   1024
#define BS_  16
#define BQ   4        // b = BS/GROUP  (number of "first" batches)
#define GB   12       // flat (g,b) pairs = (GROUP-1)*BQ
#define CG   64       // C/SG
#define HID  1024     // 4*C

typedef __attribute__((ext_vector_type(16))) _Float16 v16h;
typedef __attribute__((ext_vector_type(8)))  float    v8f;

union Frag { v16h h; uint4 u[2]; };

static __device__ __forceinline__ v8f wmma_f16(const v16h& a, const v16h& b, v8f c) {
  // D = A(16x32,f16) * B(32x16,f16) + C(16x16,f32)
  return __builtin_amdgcn_wmma_f32_16x16x32_f16(
      /*neg_a=*/false, a, /*neg_b=*/false, b,
      /*c_mod=*/(short)0, c, /*reuse_a=*/false, /*reuse_b=*/false);
}

// ---------------------------------------------------------------------------
// Prep kernels: grouped projections + f16 conversion into WMMA-friendly
// layouts. xqT[gb][n][ch], xkT[b][m][ch] (K-contiguous for NT fragments).
// ---------------------------------------------------------------------------
__global__ void prep_xqT(const float* __restrict__ x, const float* __restrict__ Wq,
                         _Float16* __restrict__ xqT) {
  int idx = blockIdx.x * blockDim.x + threadIdx.x;     // (B, n, ch), ch fastest
  if (idx >= GB * N_ * C_) return;
  int ch = idx & (C_ - 1);
  int n  = (idx >> 8) & (N_ - 1);
  int B  = idx >> 18;
  int s = ch >> 6, o = ch & 63;
  const float* col = x + ((size_t)(BQ + B) * C_ + s * CG) * N_ + n;
  const float* wr  = Wq + ((size_t)s * CG + o) * CG;
  float acc = 0.f;
  #pragma unroll 8
  for (int i = 0; i < CG; ++i) acc = fmaf(wr[i], col[(size_t)i * N_], acc);
  xqT[((size_t)B * N_ + n) * C_ + ch] = (_Float16)acc;
}

__global__ void prep_xkT(const float* __restrict__ x, const float* __restrict__ Wk,
                         _Float16* __restrict__ xkT) {
  int idx = blockIdx.x * blockDim.x + threadIdx.x;     // (b, m, ch)
  if (idx >= BQ * N_ * C_) return;
  int ch = idx & (C_ - 1);
  int m  = (idx >> 8) & (N_ - 1);
  int b  = idx >> 18;
  int s = ch >> 6, o = ch & 63;
  const float* col = x + ((size_t)b * C_ + s * CG) * N_ + m;
  const float* wr  = Wk + ((size_t)s * CG + o) * CG;
  float acc = 0.f;
  #pragma unroll 8
  for (int i = 0; i < CG; ++i) acc = fmaf(wr[i], col[(size_t)i * N_], acc);
  xkT[((size_t)b * N_ + m) * C_ + ch] = (_Float16)acc;
}

__global__ void prep_xv(const float* __restrict__ x, const float* __restrict__ Wv,
                        const float* __restrict__ bv, _Float16* __restrict__ xvh) {
  int idx = blockIdx.x * blockDim.x + threadIdx.x;     // (b, c, m), m fastest
  if (idx >= BQ * C_ * N_) return;
  int m = idx & (N_ - 1);
  int c = (idx >> 10) & (C_ - 1);
  int b = idx >> 18;
  const float* col = x + (size_t)b * C_ * N_ + m;
  const float* wr  = Wv + (size_t)c * C_;
  float acc = bv[c];
  #pragma unroll 8
  for (int k = 0; k < C_; ++k) acc = fmaf(wr[k], col[(size_t)k * N_], acc);
  xvh[idx] = (_Float16)acc;
}

__global__ void cvt_h(const float* __restrict__ in, _Float16* __restrict__ out, int n) {
  int i = blockIdx.x * blockDim.x + threadIdx.x;
  if (i < n) out[i] = (_Float16)in[i];
}

// ---------------------------------------------------------------------------
// energy + softmax fused: one block = 16 query rows x 1024 keys, K=256.
// 8 waves (256 threads); wave w owns columns [w*128, w*128+128) -> 8 WMMA
// n-tiles (acc = 64 VGPRs/lane: fits the register budget with no spills).
// Energy strip parked in LDS (64KB of the 320KB WGP LDS), softmax in-block,
// attn^T written in f16 ([gb][m][n] so the x_r GEMM gets K-contiguous B).
// ---------------------------------------------------------------------------
__global__ void __launch_bounds__(256)
energy_softmax(const _Float16* __restrict__ xqT,
               const _Float16* __restrict__ xkT,
               _Float16* __restrict__ attnT) {
  extern __shared__ char smem[];
  _Float16* aT    = (_Float16*)smem;                            // 16*256 f16 = 8KB
  float*    eT    = (float*)(smem + 16 * C_ * 2);               // 16*1024 f32 = 64KB
  float*    red   = (float*)(smem + 16 * C_ * 2 + 16 * N_ * 4); // 16*16
  float*    rstat = red + 16 * 16;                              // 16

  const int tid = threadIdx.x;
  const int gb  = blockIdx.y;
  const int b   = gb & 3;
  const int n0  = blockIdx.x * 16;

  // stage Q^T tile: rows n0..n0+15, all 256 channels
  {
    const uint4* src = (const uint4*)(xqT + ((size_t)gb * N_ + n0) * C_);
    uint4* dst = (uint4*)aT;
    for (int i = tid; i < 16 * C_ / 8; i += 256) dst[i] = src[i];
  }
  __syncthreads();

  const int lane = tid & 31;
  const int w    = tid >> 5;        // 8 waves
  const int hi16 = lane >> 4;       // 0: lanes 0-15, 1: lanes 16-31
  const int mrow = lane & 15;

  const _Float16* Bk = xkT + (size_t)b * N_ * C_;

  v8f acc[8];
  #pragma unroll
  for (int t = 0; t < 8; ++t) { v8f z = {}; acc[t] = z; }

  #pragma unroll
  for (int kc = 0; kc < C_ / 32; ++kc) {
    Frag af;
    int ab = mrow * C_ + kc * 32 + hi16 * 8;
    af.u[0] = *(const uint4*)(aT + ab);
    af.u[1] = *(const uint4*)(aT + ab + 16);
    const int bb = kc * 32 + hi16 * 8;
    #pragma unroll
    for (int mt = 0; mt < 8; ++mt) {
      int col = w * 128 + mt * 16 + mrow;
      Frag bf;
      const _Float16* bp = Bk + (size_t)col * C_ + bb;
      bf.u[0] = *(const uint4*)bp;
      bf.u[1] = *(const uint4*)(bp + 16);
      acc[mt] = wmma_f16(af.h, bf.h, acc[mt]);
    }
  }
  #pragma unroll
  for (int mt = 0; mt < 8; ++mt) {
    int col = w * 128 + mt * 16 + mrow;
    #pragma unroll
    for (int j = 0; j < 8; ++j)
      eT[(j + hi16 * 8) * N_ + col] = acc[mt][j];   // C layout: M = j + 8*(lane>=16)
  }
  __syncthreads();

  // softmax over m: 16 threads per row
  const int r = tid >> 4, q = tid & 15;
  float mx = -3.4e38f;
  for (int m = q; m < N_; m += 16) mx = fmaxf(mx, eT[r * N_ + m]);
  red[r * 16 + q] = mx;
  __syncthreads();
  if (q == 0) {
    float v = red[r * 16];
    #pragma unroll
    for (int i = 1; i < 16; ++i) v = fmaxf(v, red[r * 16 + i]);
    rstat[r] = v;
  }
  __syncthreads();
  const float rmax = rstat[r];
  float s = 0.f;
  for (int m = q; m < N_; m += 16) {
    float e = __expf(eT[r * N_ + m] - rmax);
    eT[r * N_ + m] = e;
    s += e;
  }
  __syncthreads();
  red[r * 16 + q] = s;
  __syncthreads();
  if (q == 0) {
    float v = 0.f;
    #pragma unroll
    for (int i = 0; i < 16; ++i) v += red[r * 16 + i];
    rstat[r] = 1.f / v;
  }
  __syncthreads();
  const float inv = rstat[r];
  _Float16* out = attnT + (size_t)gb * N_ * N_;
  for (int m = q; m < N_; m += 16)
    out[(size_t)m * N_ + (n0 + r)] = (_Float16)(eT[r * N_ + m] * inv);
}

// ---------------------------------------------------------------------------
// Generic NT WMMA GEMM: C[M,N] = sum_ib A_ib[M,K] * B_ib[N,K]^T
// Block = 128 threads (4 waves) -> 16 x 256 output tile.
// Epilogue: scale, per-column bias, fp32 residual, ReLU, f32 or f16 out.
// ---------------------------------------------------------------------------
struct AccSpec { int n; long aOff[3]; long bOff[3]; };

__global__ void __launch_bounds__(128)
gemm_nt(const _Float16* __restrict__ A, const _Float16* __restrict__ B,
        int K, int N, AccSpec spec, float scale,
        const float* __restrict__ bias,
        const float* __restrict__ resid, int doRelu,
        float* __restrict__ outF, _Float16* __restrict__ outH) {
  __shared__ _Float16 aT[16 * 1024];                  // 32KB, supports K<=1024
  const int tid  = threadIdx.x;
  const int lane = tid & 31;
  const int w    = tid >> 5;
  const int hi16 = lane >> 4;
  const int mrow = lane & 15;
  const int m0      = blockIdx.x * 16;
  const int colBase = blockIdx.y * 256 + w * 64;

  v8f acc[4];
  #pragma unroll
  for (int t = 0; t < 4; ++t) { v8f z = {}; acc[t] = z; }

  const int vecPerTile = 16 * K / 8;
  for (int ib = 0; ib < spec.n; ++ib) {
    const _Float16* Ae = A + spec.aOff[ib];
    const _Float16* Be = B + spec.bOff[ib];
    if (ib) __syncthreads();                          // retire readers before reload
    {
      const uint4* src = (const uint4*)(Ae + (size_t)m0 * K);
      uint4* dst = (uint4*)aT;
      for (int i = tid; i < vecPerTile; i += 128) dst[i] = src[i];
    }
    __syncthreads();
    for (int kc = 0; kc < K / 32; ++kc) {
      Frag af;
      int ab = mrow * K + kc * 32 + hi16 * 8;
      af.u[0] = *(const uint4*)(aT + ab);
      af.u[1] = *(const uint4*)(aT + ab + 16);
      const int bb = kc * 32 + hi16 * 8;
      #pragma unroll
      for (int nt = 0; nt < 4; ++nt) {
        const _Float16* bp = Be + (size_t)(colBase + nt * 16 + mrow) * K + bb;
        Frag bf;
        bf.u[0] = *(const uint4*)bp;
        bf.u[1] = *(const uint4*)(bp + 16);
        acc[nt] = wmma_f16(af.h, bf.h, acc[nt]);
      }
    }
  }

  #pragma unroll
  for (int nt = 0; nt < 4; ++nt) {
    int col = colBase + nt * 16 + mrow;
    float bc = bias ? bias[col] : 0.f;
    #pragma unroll
    for (int j = 0; j < 8; ++j) {
      int row = m0 + j + hi16 * 8;
      float v = acc[nt][j] * scale + bc;
      size_t oi = (size_t)row * N + col;
      if (resid) v += resid[oi];
      if (doRelu) v = fmaxf(v, 0.f);
      if (outF) outF[oi] = v;
      else      outH[oi] = (_Float16)v;
    }
  }
}

// ---------------------------------------------------------------------------
// LayerNorm kernels
// ---------------------------------------------------------------------------
__global__ void ln1_kernel(const float* __restrict__ h0, const float* __restrict__ g,
                           const float* __restrict__ be, float* __restrict__ hln,
                           _Float16* __restrict__ hlnH) {
  int t = blockIdx.x * blockDim.x + threadIdx.x;      // (b, m)
  if (t >= BQ * N_) return;
  int b = t >> 10, m = t & (N_ - 1);
  const float* base = h0 + (size_t)b * C_ * N_ + m;   // stride N over channels
  float s = 0.f, ss = 0.f;
  for (int c = 0; c < C_; ++c) { float v = base[(size_t)c * N_]; s += v; ss += v * v; }
  float mu  = s * (1.f / C_);
  float var = ss * (1.f / C_) - mu * mu;
  float inv = rsqrtf(var + 1e-6f);
  float*    orow = hln  + (size_t)t * C_;
  _Float16* hrow = hlnH + (size_t)t * C_;
  for (int c = 0; c < C_; ++c) {
    float v = (base[(size_t)c * N_] - mu) * inv * g[c] + be[c];
    orow[c] = v;
    hrow[c] = (_Float16)v;
  }
}

__global__ void ln2_out(const float* __restrict__ h2, const float* __restrict__ g,
                        const float* __restrict__ be, float* __restrict__ out) {
  int t = blockIdx.x * blockDim.x + threadIdx.x;      // (b, m)
  if (t >= BQ * N_) return;
  int b = t >> 10, m = t & (N_ - 1);
  const float* row = h2 + (size_t)t * C_;
  float s = 0.f, ss = 0.f;
  for (int c = 0; c < C_; ++c) { float v = row[c]; s += v; ss += v * v; }
  float mu  = s * (1.f / C_);
  float var = ss * (1.f / C_) - mu * mu;
  float inv = rsqrtf(var + 1e-6f);
  for (int c = 0; c < C_; ++c) {
    float v = fmaxf((row[c] - mu) * inv * g[c] + be[c], 0.f);
    out[(size_t)b * C_ * N_ + (size_t)c * N_ + m] = v;   // transpose back to (b,c,n)
  }
}

// ---------------------------------------------------------------------------
// Host side
// ---------------------------------------------------------------------------
extern "C" void kernel_launch(void* const* d_in, const int* in_sizes, int n_in,
                              void* d_out, int out_size, void* d_ws, size_t ws_size,
                              hipStream_t stream) {
  (void)in_sizes; (void)n_in; (void)out_size; (void)ws_size;
  const float* x   = (const float*)d_in[0];
  const float* Wq  = (const float*)d_in[1];
  const float* Wk  = (const float*)d_in[2];
  const float* Wv  = (const float*)d_in[3];
  const float* bv  = (const float*)d_in[4];
  const float* g1  = (const float*)d_in[5];
  const float* be1 = (const float*)d_in[6];
  const float* W1  = (const float*)d_in[7];
  const float* b1  = (const float*)d_in[8];
  const float* W2  = (const float*)d_in[9];
  const float* b2  = (const float*)d_in[10];
  const float* g2  = (const float*)d_in[11];
  const float* be2 = (const float*)d_in[12];

  char* ws = (char*)d_ws;
  size_t off = 0;
  auto alloc = [&](size_t bytes) -> char* {
    char* p = ws + off; off += (bytes + 255) & ~(size_t)255; return p;
  };
  _Float16* xqT   = (_Float16*)alloc((size_t)GB * N_ * C_ * 2);   // Q^T  [gb][n][ch]
  _Float16* xkT   = (_Float16*)alloc((size_t)BQ * N_ * C_ * 2);   // K^T  [b][m][ch]
  _Float16* xvh   = (_Float16*)alloc((size_t)BQ * C_ * N_ * 2);   // V    [b][c][m]
  _Float16* attnT = (_Float16*)alloc((size_t)GB * N_ * N_ * 2);   // attn^T [gb][m][n]
  float*    h0    = (float*)  alloc((size_t)BQ * C_ * N_ * 4);    // first + x_r
  float*    hln   = (float*)  alloc((size_t)BQ * N_ * C_ * 4);    // LN1 out (b,m,c)
  _Float16* hlnH  = (_Float16*)alloc((size_t)BQ * N_ * C_ * 2);
  _Float16* w1h   = (_Float16*)alloc((size_t)HID * C_ * 2);
  _Float16* w2h   = (_Float16*)alloc((size_t)C_ * HID * 2);
  _Float16* t16   = (_Float16*)alloc((size_t)BQ * N_ * HID * 2);  // relu(h@W1^T+b1)
  float*    h2    = (float*)  alloc((size_t)BQ * N_ * C_ * 4);    // pre-LN2

  prep_xqT<<<GB * N_ * C_ / 256, 256, 0, stream>>>(x, Wq, xqT);
  prep_xkT<<<BQ * N_ * C_ / 256, 256, 0, stream>>>(x, Wk, xkT);
  prep_xv <<<BQ * C_ * N_ / 256, 256, 0, stream>>>(x, Wv, bv, xvh);
  cvt_h<<<HID * C_ / 256, 256, 0, stream>>>(W1, w1h, HID * C_);
  cvt_h<<<C_ * HID / 256, 256, 0, stream>>>(W2, w2h, C_ * HID);

  size_t esmem = 16 * C_ * 2 + 16 * N_ * 4 + (16 * 16 + 16) * 4; // ~74.3KB of 320KB LDS
  energy_softmax<<<dim3(N_ / 16, GB), 256, esmem, stream>>>(xqT, xkT, attnT);

  // x_r (+ 1/48 scale + residual "first") for each output batch B.
  // Reference's reshape(b, group-1, ...).mean averages flat gb in {3B,3B+1,3B+2},
  // pairing xv[gb&3] with attn[gb].
  for (int B = 0; B < BQ; ++B) {
    AccSpec sp; sp.n = 3;
    for (int j = 0; j < 3; ++j) {
      int gb = 3 * B + j;
      sp.aOff[j] = (long)(gb & 3) * C_ * N_;
      sp.bOff[j] = (long)gb * N_ * N_;
    }
    gemm_nt<<<dim3(C_ / 16, N_ / 256), 128, 0, stream>>>(
        xvh, attnT, N_, N_, sp, 1.f / 48.f, nullptr,
        x + (size_t)B * C_ * N_, 0,
        h0 + (size_t)B * C_ * N_, nullptr);
  }

  ln1_kernel<<<BQ * N_ / 256, 256, 0, stream>>>(h0, g1, be1, hln, hlnH);

  AccSpec one; one.n = 1;
  one.aOff[0] = one.aOff[1] = one.aOff[2] = 0;
  one.bOff[0] = one.bOff[1] = one.bOff[2] = 0;

  // t = relu(hln @ W1^T + b1), f16 out
  gemm_nt<<<dim3(BQ * N_ / 16, HID / 256), 128, 0, stream>>>(
      hlnH, w1h, C_, HID, one, 1.f, b1, nullptr, 1, nullptr, t16);
  // h2 = hln + t @ W2^T + b2, f32 out
  gemm_nt<<<dim3(BQ * N_ / 16, C_ / 256), 128, 0, stream>>>(
      t16, w2h, HID, C_, one, 1.f, b2, hln, 0, h2, nullptr);

  ln2_out<<<BQ * N_ / 256, 256, 0, stream>>>(h2, g2, be2, (float*)d_out);

  // batches 4..15 pass through
  hipMemcpyAsync((float*)d_out + (size_t)BQ * C_ * N_,
                 x + (size_t)BQ * C_ * N_,
                 (size_t)(BS_ - BQ) * C_ * N_ * sizeof(float),
                 hipMemcpyDeviceToDevice, stream);
}
